// FermiNet_17643725652500
// MI455X (gfx1250) — compile-verified, matrix operand order checked by reading, not run on hardware
//
#include <hip/hip_runtime.h>
#include <hip/hip_bf16.h>
#include <math.h>

// ---------------------------------------------------------------------------
// FermiNet forward for gfx1250 (MI455X).  One workgroup per batch element,
// all activations in LDS, ALL matmuls (unmixed, mixed, 2e->1e, two-e,
// orbitals) on v_wmma_f32_16x16x32_bf16.
// ---------------------------------------------------------------------------

typedef __attribute__((ext_vector_type(16))) __bf16 bf16x16;
typedef __attribute__((ext_vector_type(8)))  float  floatx8;

struct FNParams {
  const float* elec;       // (B,16,3)
  const float* ion;        // (4,3)
  const float* b1[4];      // one-e bias (256)
  const float* b2[4];      // two-e bias (32)
  const float* orb_bias;   // (2,128)
  const float* env_decay;  // (2,4,128,3,3)
  const float* env_ion;    // (2,4,128)
  const __bf16* un[4];     // staged bf16 [256][Kpad]   (unmixed one-e weights)
  const __bf16* mx[4];     // staged bf16 [256][2*Fin]  (mixed weights)
  const __bf16* t21[4];    // staged bf16 [256][Kpad]   (2e->1e weights)
  const __bf16* e2[4];     // staged bf16 [32][32]      (two-e weights)
  const __bf16* orb[2];    // staged bf16 [128][256]    (orbital kernels)
  float* out;              // (B,)
};

__device__ __forceinline__ bf16x16 bz16() {
  bf16x16 z;
#pragma unroll
  for (int i = 0; i < 16; ++i) z[i] = (__bf16)0.0f;
  return z;
}
__device__ __forceinline__ floatx8 fz8() {
  floatx8 z;
#pragma unroll
  for (int i = 0; i < 8; ++i) z[i] = 0.0f;
  return z;
}

// 16-bit A-matrix striping (ISA 7.12.2): per lane, elements 0..7 and 8..15
// are two contiguous 8-element K runs -> compiler vectorizes to b128 loads.
__device__ __forceinline__ bf16x16 frag_ld(const __bf16* p, int row, int stride,
                                           int kb, int hi) {
  bf16x16 f;
#pragma unroll
  for (int e = 0; e < 16; ++e) {
    int r = e >> 1;
    int k = ((r < 4) ? 2 * r : 16 + 2 * (r - 4)) + (hi ? 8 : 0) + (e & 1) + kb;
    f[e] = p[row * stride + k];
  }
  return f;
}

__device__ __forceinline__ floatx8 wmma_bf16(bf16x16 a, bf16x16 b, floatx8 c) {
  return __builtin_amdgcn_wmma_f32_16x16x32_bf16(false, a, false, b, (short)0,
                                                 c, false, false);
}

// One residual layer (one-e + two-e).  Templated so K-loops fully unroll
// into static WMMA chains.
template <int FIN, int KUN, int K21, int C2, int SKIP>
__device__ void one_layer(int t, int m, int hi, int wave,
                          float (*x1)[256], __bf16 (*x1b)[256],
                          __bf16 (*x2b)[32], __bf16 (*cat2b)[64],
                          float (*means1)[256], __bf16 (*amx)[512],
                          const __bf16* UN, const __bf16* MX, const __bf16* T21,
                          const __bf16* E2, const float* B1, const float* B2) {
  // ---- phase A1: stage bf16 copies + spin means -------------------------
  for (int idx = t; idx < 16 * KUN; idx += 256) {
    int e = idx / KUN, f = idx % KUN;
    x1b[e][f] = (f < FIN) ? (__bf16)x1[e][f] : (__bf16)0.0f;
  }
  if (t < FIN) {
    float s0 = 0.0f, s1 = 0.0f;
    for (int e = 0; e < 8; ++e) { s0 += x1[e][t]; s1 += x1[8 + e][t]; }
    means1[0][t] = s0 * 0.125f;
    means1[1][t] = s1 * 0.125f;
  }
  // per-electron 2e means, spin-ordered concat, zero-padded to 64 cols
  for (int idx = t; idx < 1024; idx += 256) {
    int i = idx >> 6, c = idx & 63, s = i >> 3;
    __bf16 v = (__bf16)0.0f;
    if (c < 2 * C2) {
      int jspin = (c < C2) ? s : (1 - s);
      int cc = (c < C2) ? c : (c - C2);
      float sum = 0.0f;
      for (int e2 = 0; e2 < 8; ++e2) sum += (float)x2b[i * 16 + jspin * 8 + e2][cc];
      v = (__bf16)(sum * 0.125f);
    }
    cat2b[i][c] = v;
  }
  __syncthreads();
  // ---- phase A2: mixed-stream A matrix (2 rows, spin-ordered concat) ----
  for (int idx = t; idx < 1024; idx += 256) {
    int s = idx >> 9, f = idx & 511;
    __bf16 v = (__bf16)0.0f;
    if (f < FIN)            v = (__bf16)means1[s][f];
    else if (f < 2 * FIN)   v = (__bf16)means1[1 - s][f - FIN];
    amx[s][f] = v;
  }
  __syncthreads();
  // ---- phase B: one-e GEMM: unmixed + mixed + 2e->1e, all WMMA ----------
  float vals[2][8];
  for (int tt = 0; tt < 2; ++tt) {
    int tn = wave * 2 + tt;
    floatx8 acc = fz8();
#pragma unroll
    for (int kb = 0; kb < KUN; kb += 32) {
      bf16x16 a = frag_ld(&x1b[0][0], m, 256, kb, hi);
      bf16x16 w = frag_ld(UN, tn * 16 + m, KUN, kb, hi);
      acc = wmma_bf16(a, w, acc);
    }
#pragma unroll
    for (int kb = 0; kb < 2 * FIN; kb += 32) {   // mixed: A row = spin(m)
      bf16x16 a = frag_ld(&amx[0][0], m >> 3, 512, kb, hi);
      bf16x16 w = frag_ld(MX, tn * 16 + m, 2 * FIN, kb, hi);
      acc = wmma_bf16(a, w, acc);
    }
#pragma unroll
    for (int kb = 0; kb < K21; kb += 32) {
      bf16x16 a = frag_ld(&cat2b[0][0], m, 64, kb, hi);
      bf16x16 w = frag_ld(T21, tn * 16 + m, K21, kb, hi);
      acc = wmma_bf16(a, w, acc);
    }
    int col = tn * 16 + m;
#pragma unroll
    for (int v = 0; v < 8; ++v) {
      int row = v + 8 * hi;
      float val = acc[v] + B1[col];
      val = tanhf(val);
      if (SKIP) val += x1[row][col];
      vals[tt][v] = val;
    }
  }
  __syncthreads();
  for (int tt = 0; tt < 2; ++tt) {
    int col = (wave * 2 + tt) * 16 + m;
#pragma unroll
    for (int v = 0; v < 8; ++v) x1[v + 8 * hi][col] = vals[tt][v];
  }
  __syncthreads();
  // ---- phase C: two-e GEMM (256 x 32 @ 32 x 32) -------------------------
  float v2[4][8];
  for (int q = 0; q < 4; ++q) {
    int tm = wave * 2 + (q >> 1), tn = q & 1;
    floatx8 acc = fz8();
    bf16x16 a = frag_ld(&x2b[0][0], tm * 16 + m, 32, 0, hi);
    bf16x16 w = frag_ld(E2, tn * 16 + m, 32, 0, hi);
    acc = wmma_bf16(a, w, acc);
    int col = tn * 16 + m;
#pragma unroll
    for (int v = 0; v < 8; ++v) {
      int rg = tm * 16 + v + 8 * hi;
      float val = tanhf(acc[v] + B2[col]);
      if (SKIP) val += (float)x2b[rg][col];
      v2[q][v] = val;
    }
  }
  __syncthreads();
  for (int q = 0; q < 4; ++q) {
    int tm = wave * 2 + (q >> 1), tn = q & 1, col = tn * 16 + m;
#pragma unroll
    for (int v = 0; v < 8; ++v) x2b[tm * 16 + v + 8 * hi][col] = (__bf16)v2[q][v];
  }
  __syncthreads();
}

__global__ __launch_bounds__(256) void ferminet_fwd(FNParams P) {
  __shared__ float  x1[16][256];      // one-e activations (f32)
  __shared__ __bf16 x1b[16][256];     // bf16 staging for WMMA A
  __shared__ __bf16 x2b[256][32];     // two-e activations (bf16)
  __shared__ __bf16 cat2b[16][64];    // 2e-mean concat A matrix
  __shared__ __bf16 amx[2][512];      // mixed-stream A matrix (per spin)
  __shared__ float  means1[2][256];
  __shared__ float  ep[16][4];
  __shared__ float  rei[16][4][4];
  __shared__ float  envb[2][8][128];
  __shared__ float  orbs[2][16][8][8];
  __shared__ float  s_ld[2][16], s_sg[2][16], s_tl[16], s_ts[16];

  const int b = blockIdx.x;
  const int t = threadIdx.x;
  const int lane = t & 31, wave = t >> 5;
  const int m = lane & 15, hi = (lane >> 4) & 1;

  // ---- features ---------------------------------------------------------
  if (t < 48) { int e = t / 3, d = t % 3; ep[e][d] = P.elec[(b * 16 + e) * 3 + d]; }
  __syncthreads();
  if (t < 64) {
    int e = t >> 2, a = t & 3;
    float rx = ep[e][0] - P.ion[a * 3 + 0];
    float ry = ep[e][1] - P.ion[a * 3 + 1];
    float rz = ep[e][2] - P.ion[a * 3 + 2];
    float nr = sqrtf(rx * rx + ry * ry + rz * rz);
    rei[e][a][0] = rx; rei[e][a][1] = ry; rei[e][a][2] = rz; rei[e][a][3] = nr;
    x1[e][a * 4 + 0] = rx; x1[e][a * 4 + 1] = ry;
    x1[e][a * 4 + 2] = rz; x1[e][a * 4 + 3] = nr;
  }
  {
    int i = t >> 4, j = t & 15;
    float rx = ep[i][0] - ep[j][0];
    float ry = ep[i][1] - ep[j][1];
    float rz = ep[i][2] - ep[j][2];
    float nr = (i == j) ? 0.0f : sqrtf(rx * rx + ry * ry + rz * rz);
    x2b[t][0] = (__bf16)rx; x2b[t][1] = (__bf16)ry;
    x2b[t][2] = (__bf16)rz; x2b[t][3] = (__bf16)nr;
    for (int c = 4; c < 32; ++c) x2b[t][c] = (__bf16)0.0f;
  }
  __syncthreads();
  // ---- envelope (VALU) --------------------------------------------------
  for (int idx = t; idx < 2048; idx += 256) {
    int s = idx >> 10, rem = idx & 1023, i = rem >> 7, k = rem & 127;
    float acc = 0.0f;
    for (int a = 0; a < 4; ++a) {
      const float* Dm = P.env_decay + ((size_t)(s * 4 + a) * 128 + k) * 9;
      float r0 = rei[s * 8 + i][a][0];
      float r1 = rei[s * 8 + i][a][1];
      float r2 = rei[s * 8 + i][a][2];
      float a0 = Dm[0] * r0 + Dm[1] * r1 + Dm[2] * r2;
      float a1 = Dm[3] * r0 + Dm[4] * r1 + Dm[5] * r2;
      float a2 = Dm[6] * r0 + Dm[7] * r1 + Dm[8] * r2;
      float dist = sqrtf(a0 * a0 + a1 * a1 + a2 * a2);
      acc += expf(-dist) * P.env_ion[(s * 4 + a) * 128 + k];
    }
    envb[s][i][k] = acc;
  }
  __syncthreads();

  // ---- 4 residual layers ------------------------------------------------
  one_layer<16, 32, 32, 4, 0>(t, m, hi, wave, x1, x1b, x2b, cat2b, means1, amx,
                              P.un[0], P.mx[0], P.t21[0], P.e2[0],
                              P.b1[0], P.b2[0]);
  for (int l = 1; l < 4; ++l)
    one_layer<256, 256, 64, 32, 1>(t, m, hi, wave, x1, x1b, x2b, cat2b, means1,
                                   amx, P.un[l], P.mx[l], P.t21[l], P.e2[l],
                                   P.b1[l], P.b2[l]);

  // ---- orbital GEMM per spin (8x256 @ 256x128, M padded to 16) ----------
  for (int idx = t; idx < 4096; idx += 256)
    x1b[idx >> 8][idx & 255] = (__bf16)x1[idx >> 8][idx & 255];
  __syncthreads();
  for (int s = 0; s < 2; ++s) {
    int tn = wave;   // 8 waves cover N=128
    floatx8 acc = fz8();
#pragma unroll
    for (int kb = 0; kb < 256; kb += 32) {
      bf16x16 a = (m < 8) ? frag_ld(&x1b[0][0], s * 8 + m, 256, kb, hi) : bz16();
      bf16x16 w = frag_ld(P.orb[s], tn * 16 + m, 256, kb, hi);
      acc = wmma_bf16(a, w, acc);
    }
    int o = tn * 16 + m;
    if (hi == 0) {
#pragma unroll
      for (int v = 0; v < 8; ++v) {
        float val = (acc[v] + P.orb_bias[s * 128 + o]) * envb[s][v][o];
        orbs[s][o >> 3][v][o & 7] = val;
      }
    }
  }
  __syncthreads();

  // ---- slogdet: 32 independent 8x8 LU (one per lane) --------------------
  if (t < 32) {
    int s = t >> 4, d = t & 15;
    float A[8][8];
    for (int i = 0; i < 8; ++i)
      for (int j = 0; j < 8; ++j) A[i][j] = orbs[s][d][i][j];
    float sg = 1.0f, ldet = 0.0f;
    for (int c = 0; c < 8; ++c) {
      int p = c; float best = fabsf(A[c][c]);
      for (int r = c + 1; r < 8; ++r) {
        float v = fabsf(A[r][c]);
        if (v > best) { best = v; p = r; }
      }
      if (p != c) {
        for (int j = 0; j < 8; ++j) {
          float tmp = A[c][j]; A[c][j] = A[p][j]; A[p][j] = tmp;
        }
        sg = -sg;
      }
      float piv = A[c][c];
      if (piv < 0.0f) sg = -sg;
      ldet += logf(fabsf(piv) + 1e-37f);
      float inv = 1.0f / piv;
      for (int r = c + 1; r < 8; ++r) {
        float fmul = A[r][c] * inv;
        for (int j = c + 1; j < 8; ++j) A[r][j] -= fmul * A[c][j];
      }
    }
    s_ld[s][d] = ldet; s_sg[s][d] = sg;
  }
  __syncthreads();
  if (t < 16) {
    s_tl[t] = s_ld[0][t] + s_ld[1][t];
    s_ts[t] = s_sg[0][t] * s_sg[1][t];
  }
  __syncthreads();
  if (t == 0) {
    float mx = s_tl[0];
    for (int d = 1; d < 16; ++d) mx = fmaxf(mx, s_tl[d]);
    float sum = 0.0f;
    for (int d = 0; d < 16; ++d) sum += s_ts[d] * expf(s_tl[d] - mx);
    P.out[b] = mx + logf(fabsf(sum) + 1e-37f);
  }
}

// f32 (K,N) row-major  ->  bf16 (N, Kpad) with zero K-padding
__global__ void conv_w_bf16(const float* src, __bf16* dst, int K, int N, int Kpad) {
  int idx = blockIdx.x * 256 + threadIdx.x;
  if (idx >= N * Kpad) return;
  int n = idx / Kpad, k = idx % Kpad;
  dst[idx] = (k < K) ? (__bf16)src[k * N + n] : (__bf16)0.0f;
}

extern "C" void kernel_launch(void* const* d_in, const int* in_sizes, int n_in,
                              void* d_out, int out_size, void* d_ws, size_t ws_size,
                              hipStream_t stream) {
  (void)n_in; (void)ws_size; (void)out_size;
  const float* elec      = (const float*)d_in[0];
  const float* ion       = (const float*)d_in[1];
  const float* w0_un     = (const float*)d_in[2];
  const float* w0_mixed  = (const float*)d_in[3];
  const float* w0_2e1e   = (const float*)d_in[4];
  const float* b0_1e     = (const float*)d_in[5];
  const float* w0_2e     = (const float*)d_in[6];
  const float* b0_2e     = (const float*)d_in[7];
  const float* w_un      = (const float*)d_in[8];
  const float* w_mixed   = (const float*)d_in[9];
  const float* w_2e1e    = (const float*)d_in[10];
  const float* b_1e      = (const float*)d_in[11];
  const float* w_2e      = (const float*)d_in[12];
  const float* b_2e      = (const float*)d_in[13];
  const float* orb_k     = (const float*)d_in[14];
  const float* orb_bias  = (const float*)d_in[15];
  const float* env_decay = (const float*)d_in[16];
  const float* env_ion   = (const float*)d_in[17];

  const int B = in_sizes[0] / (16 * 3);

  // workspace bf16 staging layout
  size_t off = 0;
  __bf16* base = (__bf16*)d_ws;
  auto take = [&](size_t elems) { __bf16* p = base + off; off += elems; return p; };
  __bf16* UN[4];  UN[0]  = take(256 * 32);
  for (int l = 1; l < 4; ++l) UN[l]  = take(256 * 256);
  __bf16* MX[4];  MX[0]  = take(256 * 32);
  for (int l = 1; l < 4; ++l) MX[l]  = take(256 * 512);
  __bf16* T21[4]; T21[0] = take(256 * 32);
  for (int l = 1; l < 4; ++l) T21[l] = take(256 * 64);
  __bf16* E2[4];  E2[0]  = take(32 * 32);
  for (int l = 1; l < 4; ++l) E2[l]  = take(32 * 32);
  __bf16* ORB[2];
  for (int s = 0; s < 2; ++s) ORB[s] = take(128 * 256);

  auto cv = [&](const float* s, __bf16* d, int K, int N, int Kp) {
    int tot = N * Kp;
    conv_w_bf16<<<(tot + 255) / 256, 256, 0, stream>>>(s, d, K, N, Kp);
  };
  cv(w0_un, UN[0], 16, 256, 32);
  for (int l = 1; l < 4; ++l) cv(w_un + (size_t)(l - 1) * 256 * 256, UN[l], 256, 256, 256);
  cv(w0_mixed, MX[0], 32, 256, 32);
  for (int l = 1; l < 4; ++l) cv(w_mixed + (size_t)(l - 1) * 512 * 256, MX[l], 512, 256, 512);
  cv(w0_2e1e, T21[0], 8, 256, 32);
  for (int l = 1; l < 4; ++l) cv(w_2e1e + (size_t)(l - 1) * 64 * 256, T21[l], 64, 256, 64);
  cv(w0_2e, E2[0], 4, 32, 32);
  for (int l = 1; l < 4; ++l) cv(w_2e + (size_t)(l - 1) * 32 * 32, E2[l], 32, 32, 32);
  for (int s = 0; s < 2; ++s) cv(orb_k + (size_t)s * 256 * 128, ORB[s], 256, 128, 256);

  FNParams P;
  P.elec = elec; P.ion = ion;
  P.b1[0] = b0_1e; P.b2[0] = b0_2e;
  for (int l = 1; l < 4; ++l) {
    P.b1[l] = b_1e + (size_t)(l - 1) * 256;
    P.b2[l] = b_2e + (size_t)(l - 1) * 32;
  }
  P.orb_bias = orb_bias; P.env_decay = env_decay; P.env_ion = env_ion;
  for (int l = 0; l < 4; ++l) {
    P.un[l] = UN[l]; P.mx[l] = MX[l]; P.t21[l] = T21[l]; P.e2[l] = E2[l];
  }
  P.orb[0] = ORB[0]; P.orb[1] = ORB[1];
  P.out = (float*)d_out;

  ferminet_fwd<<<B, 256, 0, stream>>>(P);
}
